// MultiOrganismLinear_42838003810507
// MI455X (gfx1250) — matrix operand order repaired and checked by reading, not compiled, not run on hardware
//
#include <hip/hip_runtime.h>

// bf16 WMMA fragment/accumulator vector types (CDNA5 wave32 layouts)
typedef __attribute__((ext_vector_type(16))) __bf16 v16bf;
typedef __attribute__((ext_vector_type(8)))  float  v8f;
typedef __attribute__((ext_vector_type(2)))  __bf16 bf16x2;
typedef __attribute__((ext_vector_type(2)))  float  f32x2;

// Problem sizes fixed by the reference
constexpr int BATCH = 8;
constexpr int SEQ   = 4096;
constexpr int IN    = 1536;
constexpr int OUTD  = 3072;

// Tiling
constexpr int BM = 128, BN = 128, BK = 32;
constexpr int KSTEPS   = IN / BK;        // 48
constexpr int A_STRIDE = 80;             // bytes/row in LDS: 32 bf16 (64B) + 16B pad
constexpr int B_STRIDE = 80;             // bytes per N-column row: 16 bf16-pairs (64B) + pad
constexpr int A_TILE   = BM * A_STRIDE;  // 10240 B
constexpr int B_TILE   = BN * B_STRIDE;  // 10240 B
constexpr int BUF      = A_TILE + B_TILE;// 20480 B per double-buffer phase

union FragAB { uint4 q[2]; v16bf v; };

// One packed convert: v_cvt_pk_bf16_f32 dst, a, b
__device__ __forceinline__ unsigned pack_bf16(float a, float b) {
  union { bf16x2 h; unsigned u; } r;
#if __has_builtin(__builtin_amdgcn_cvt_pk_bf16_f32)
  r.h = __builtin_amdgcn_cvt_pk_bf16_f32(a, b);
#else
  f32x2 f = {a, b};
  r.h = __builtin_convertvector(f, bf16x2);
#endif
  return r.u;
}

__global__ __launch_bounds__(256)
void mol_wmma_kernel(const float* __restrict__ x,
                     const void* __restrict__ orgp, int orgIs64,
                     const float* __restrict__ weight,
                     const float* __restrict__ bias,
                     float* __restrict__ out)
{
  __shared__ __align__(16) unsigned char lds[2 * BUF];

  const int t  = threadIdx.x;
  const int b  = blockIdx.z;
  const int m0 = blockIdx.y * BM;
  const int n0 = blockIdx.x * BN;

  const int o = orgIs64 ? (int)((const long long*)orgp)[b]
                        : ((const int*)orgp)[b];

  const float* xb = x      + (size_t)b * SEQ * IN + (size_t)m0 * IN;
  const float* wb = weight + (size_t)o * IN * OUTD;

  const int lane  = t & 31;
  const int ln    = lane & 15;   // M (A) / N (B) index within 16
  const int kh    = lane >> 4;   // K-half selector per ISA layout
  const int wave  = t >> 5;
  const int waveM = wave >> 2;   // 0..1 : 64 rows each
  const int waveN = wave & 3;    // 0..3 : 32 cols each

  // ---- Hoisted per-thread addressing (loop only does constant increments) ----
  const float* aP[4];            // A staging global pointers
  int aWoff[4];                  // A staging LDS byte offsets
#pragma unroll
  for (int i = 0; i < 4; ++i) {
    const int u   = i * 256 + t;
    const int row = u >> 3;      // 8 float4 per 32-float row
    const int c4  = u & 7;
    aP[i]    = xb + (size_t)row * IN + c4 * 4;
    aWoff[i] = row * A_STRIDE + c4 * 8;
  }

  const float* bP[2];            // B staging global pointers (even K row; odd via +OUTD imm)
  int bWoff[2];                  // B staging LDS byte offsets (j*B_STRIDE added as imm)
#pragma unroll
  for (int i = 0; i < 2; ++i) {
    const int u  = i * 256 + t;
    const int kp = u >> 5;       // k-pair 0..15
    const int nc = u & 31;       // 4-wide N chunk
    bP[i]    = wb + (size_t)(2 * kp) * OUTD + n0 + nc * 4;
    bWoff[i] = A_TILE + (nc * 4) * B_STRIDE + kp * 4;
  }

  // Fragment read base offsets (i*16*A_STRIDE / j*16*B_STRIDE folded as immediates)
  const int aRoff = (waveM * 64 + ln) * A_STRIDE + kh * 16;
  const int bRoff = A_TILE + (waveN * 32 + ln) * B_STRIDE + kh * 16;

  v8f acc[4][2] = {};

  float4 aReg[4];
  float4 bRegLo[2], bRegHi[2];

  auto globalLoad = [&]() {
#pragma unroll
    for (int i = 0; i < 4; ++i)
      aReg[i] = *(const float4*)(aP[i]);
#pragma unroll
    for (int i = 0; i < 2; ++i) {
      bRegLo[i] = *(const float4*)(bP[i]);
      bRegHi[i] = *(const float4*)(bP[i] + OUTD);   // +12288B instruction offset
    }
  };

  globalLoad();   // tile ks=0 into registers

#pragma unroll 2
  for (int ks = 0; ks < KSTEPS; ++ks) {
    const int curBase = (ks & 1) * BUF;     // constant-folds under unroll-by-2
    __syncthreads();                        // buf[cur] no longer being read

    // registers -> bf16 -> LDS (one v_cvt_pk_bf16_f32 per pair)
#pragma unroll
    for (int i = 0; i < 4; ++i) {
      uint2 pv;
      pv.x = pack_bf16(aReg[i].x, aReg[i].y);
      pv.y = pack_bf16(aReg[i].z, aReg[i].w);
      *(uint2*)(lds + curBase + aWoff[i]) = pv;
    }
#pragma unroll
    for (int i = 0; i < 2; ++i) {
      const float lo[4] = {bRegLo[i].x, bRegLo[i].y, bRegLo[i].z, bRegLo[i].w};
      const float hi[4] = {bRegHi[i].x, bRegHi[i].y, bRegHi[i].z, bRegHi[i].w};
#pragma unroll
      for (int j = 0; j < 4; ++j)
        *(unsigned*)(lds + curBase + bWoff[i] + j * B_STRIDE) =
            pack_bf16(lo[j], hi[j]);
    }

    __syncthreads();                        // tile visible to all waves

    // prefetch next K-step into registers (overlaps HBM with WMMA)
    if (ks + 1 < KSTEPS) {
#pragma unroll
      for (int i = 0; i < 4; ++i) aP[i] += BK;                  // +128 B
#pragma unroll
      for (int i = 0; i < 2; ++i) bP[i] += (size_t)BK * OUTD;   // +384 KB
      globalLoad();
    }

    FragAB afr[4], bfr[2];
#pragma unroll
    for (int i = 0; i < 4; ++i) {
      // A-matrix 16x32 bf16: lane = row ln, K runs [8*kh,+8) and [16+8*kh,+8)
      const unsigned char* p = lds + curBase + aRoff + i * (16 * A_STRIDE);
      afr[i].q[0] = *(const uint4*)(p);
      afr[i].q[1] = *(const uint4*)(p + 32);
    }
#pragma unroll
    for (int j = 0; j < 2; ++j) {
      // B-matrix 32x16 bf16: lane = column ln, K runs [8*kh,+8) and [16+8*kh,+8)
      const unsigned char* p = lds + curBase + bRoff + j * (16 * B_STRIDE);
      bfr[j].q[0] = *(const uint4*)(p);
      bfr[j].q[1] = *(const uint4*)(p + 32);
    }

#pragma unroll
    for (int i = 0; i < 4; ++i)
#pragma unroll
      for (int j = 0; j < 2; ++j)
        acc[i][j] = __builtin_amdgcn_wmma_f32_16x16x32_bf16(
            false, afr[i].v, false, bfr[j].v, (short)0, acc[i][j],
            false, false);
  }

  // Epilogue. C/D layout: lanes 0-15 -> M = r, lanes 16-31 -> M = 8 + r; N = ln.
#pragma unroll
  for (int j = 0; j < 2; ++j) {
    const int col  = n0 + waveN * 32 + j * 16 + ln;
    const float bv = bias[(size_t)o * OUTD + col];
#pragma unroll
    for (int i = 0; i < 4; ++i) {
      const int rowBase = m0 + waveM * 64 + i * 16 + kh * 8;
      float* op = out + (size_t)b * SEQ * OUTD + (size_t)rowBase * OUTD + col;
#pragma unroll
      for (int r = 0; r < 8; ++r)
        op[(size_t)r * OUTD] = acc[i][j][r] + bv;
    }
  }
}

extern "C" void kernel_launch(void* const* d_in, const int* in_sizes, int n_in,
                              void* d_out, int out_size, void* d_ws, size_t ws_size,
                              hipStream_t stream) {
  const float* x      = (const float*)d_in[0];
  const void*  org    = d_in[1];                 // organism_index (int32 or int64)
  const float* weight = (const float*)d_in[2];
  const float* bias   = (const float*)d_in[3];
  float* out = (float*)d_out;

  // If the harness counts int64 elements as 32-bit words, in_sizes[1] == 16.
  const int orgIs64 = (in_sizes[1] >= 2 * BATCH) ? 1 : 0;

  dim3 grid(OUTD / BN, SEQ / BM, BATCH);
  mol_wmma_kernel<<<grid, 256, 0, stream>>>(x, org, orgIs64, weight, bias, out);
}
